// GATv2Network4View_86208583566036
// MI455X (gfx1250) — matrix operand order repaired and checked by reading, not compiled, not run on hardware
//
#include <hip/hip_runtime.h>

// ---------------------------------------------------------------------------
// GATv2 4-view network for MI455X (gfx1250).
// Big GEMMs (X@W, qkv, out-proj) run on v_wmma_f32_16x16x32_bf16 (bf16 in,
// f32 accumulate). Weight slabs are staged to LDS with the gfx1250 async
// global->LDS path (ASYNCcnt) and read back as WMMA B-fragments via ds loads.
// Edge scatter is f32-atomic (agg buffer fits in the 192MB L2).
// ---------------------------------------------------------------------------

typedef __bf16 bf16;
typedef __attribute__((ext_vector_type(16))) __bf16 v16bf;
typedef __attribute__((ext_vector_type(8)))  __bf16 v8bf;
typedef __attribute__((ext_vector_type(8)))  float  v8f;
typedef int v4i_g __attribute__((vector_size(16)));   // matches builtin param type

#define NEG_SLOPE 0.2f

#if __has_builtin(__builtin_amdgcn_global_load_async_to_lds_b128) && \
    __has_builtin(__builtin_amdgcn_s_wait_asynccnt)
#define HAVE_ASYNC_LDS 1
#else
#define HAVE_ASYNC_LDS 0
#endif

// ---------------- elementwise conversion kernels ----------------

__global__ void k_cvt_plain(const float* __restrict__ src, bf16* __restrict__ dst, long n) {
    long i = (long)blockIdx.x * blockDim.x + threadIdx.x;
    if (i < n) dst[i] = (bf16)src[i];
}

// W [K, D] f32  ->  Wt [D, K] bf16   (so GEMM's B columns are K-contiguous)
__global__ void k_cvt_wt(const float* __restrict__ W, bf16* __restrict__ Wt,
                         int K, int D, long n) {
    long i = (long)blockIdx.x * blockDim.x + threadIdx.x;
    if (i >= n) return;
    int k = (int)(i / D), d = (int)(i % D);
    Wt[(size_t)d * K + k] = (bf16)W[i];
}

// agg [bv*N+n, 256] f32 -> dst [(n*4+bv), 256] bf16  (MHA token ordering)
__global__ void k_cvt_nv(const float* __restrict__ src, bf16* __restrict__ dst,
                         int N, long n) {
    long i = (long)blockIdx.x * blockDim.x + threadIdx.x;
    if (i >= n) return;
    int d  = (int)(i & 255);
    int r  = (int)(i >> 8);
    int bv = r / N;
    int nd = r - bv * N;
    dst[((size_t)(nd * 4 + bv) << 8) + d] = (bf16)src[i];
}

__global__ void k_zero(float* __restrict__ p, long n) {
    long i = (long)blockIdx.x * blockDim.x + threadIdx.x;
    if (i < n) p[i] = 0.0f;
}

// ---------------- WMMA GEMM: C[M,Nc] = A[M,K](bf16) * Bt[Nc,K]^T + bias ----
// Block = 256 threads = 8 waves = 8 M-tiles, all sharing one 64-column weight
// slab staged in LDS (async global->LDS). Each wave computes a 16x64 strip:
// one A fragment feeds 4 WMMAs per K-step.
// Fragment layouts used here (wave32; A and B share contiguous k = 16h+s, not the ISA table order):
//  A 16x32 bf16 : lane m=l&15, kb=(l<16?0:16); VGPR0-3 = K kb..kb+7,
//                 VGPR4-7 = K kb+8..kb+15      -> two 16B global loads
//  B 32x16 bf16 : lane n=l&15, kb=(l<16?0:16); VGPR0-7 = K kb..kb+15
//                 (column n K-contiguous in slab) -> 32B LDS read
//  C 16x16 f32  : lane n=l&15, VGPR j -> row j + (l<16?0:8)

__global__ void __launch_bounds__(256)
k_wmma_gemm(const bf16* __restrict__ A, const bf16* __restrict__ Bt,
            float* __restrict__ C, const float* __restrict__ bias,
            int Mtiles, int K, int Nc) {
    __shared__ __align__(32) bf16 sB[64 * 256];     // up to 64 cols x K=256

    const int tid     = threadIdx.x;
    const int nStrips = Nc >> 6;                     // 64-column strips
    const int bn      = blockIdx.x % nStrips;
    const int bmBlk   = blockIdx.x / nStrips;

    // ---- stage weight slab: Bt rows [bn*64, bn*64+64) are contiguous ----
    const int slabBytes = 64 * K * 2;
    {
        char*       lb = (char*)sB;
        const char* gb = (const char*)(Bt + (size_t)bn * 64 * K);
#if HAVE_ASYNC_LDS
        for (int u = tid * 16; u < slabBytes; u += 256 * 16)
            __builtin_amdgcn_global_load_async_to_lds_b128(
                (__attribute__((address_space(1))) v4i_g*)((char*)gb + u),
                (__attribute__((address_space(3))) v4i_g*)(lb + u), 0, 0);
        __builtin_amdgcn_s_wait_asynccnt(0);
#else
        for (int u = tid * 16; u < slabBytes; u += 256 * 16)
            *(int4*)(lb + u) = *(const int4*)(gb + u);
#endif
    }
    __syncthreads();

    const int wv   = tid >> 5;
    const int lane = tid & 31;
    const int tm   = bmBlk * 8 + wv;
    if (tm >= Mtiles) return;                        // wave-uniform exit

    const int  aRow = tm * 16 + (lane & 15);
    const int  kbA  = (lane & 16) ? 16 : 0;
    const int  kbB  = (lane & 16) ? 16 : 0;
    const bf16* ap  = A + (size_t)aRow * K + kbA;
    const bf16* bl  = sB + (lane & 15) * K + kbB;    // + s*16*K per sub-tile

    v8f acc[4] = {};
    for (int k0 = 0; k0 < K; k0 += 32) {
        union { v16bf v; v8bf h[2]; } a;
        a.h[0] = *(const v8bf*)(ap + k0);
        a.h[1] = *(const v8bf*)(ap + k0 + 8);
        if (k0 + 32 < K) __builtin_prefetch(ap + k0 + 32, 0, 1);
        #pragma unroll
        for (int s = 0; s < 4; ++s) {
            v16bf b = *(const v16bf*)(bl + s * 16 * K + k0);   // ds_load
            acc[s] = __builtin_amdgcn_wmma_f32_16x16x32_bf16(
                         false, a.v, false, b, (short)0, acc[s], false, false);
        }
    }

    const int mBase = tm * 16 + ((lane & 16) ? 8 : 0);
    #pragma unroll
    for (int s = 0; s < 4; ++s) {
        int col  = bn * 64 + s * 16 + (lane & 15);
        float bb = bias ? bias[col] : 0.0f;
        #pragma unroll
        for (int j = 0; j < 8; ++j)
            C[(size_t)(mBase + j) * Nc + col] = acc[s][j] + bb;
    }
}

// ---------------- GAT attention scores ----------------

// s_src/s_dst[(bv*N+n)*4+h] = dot(leakyrelu(h[.., h,:]), att[h, :64 / 64:])
__global__ void k_att_scores(const float* __restrict__ Hf, const float* __restrict__ att,
                             float* __restrict__ ss, float* __restrict__ sd, long n) {
    long i = (long)blockIdx.x * blockDim.x + threadIdx.x;
    if (i >= n) return;
    int hh = (int)(i & 3);
    long r = i >> 2;                               // r = bv*N+n
    const float* hp = Hf + (r << 8) + hh * 64;
    const float* as = att + hh * 128;
    float s0 = 0.f, s1 = 0.f;
    #pragma unroll 8
    for (int f = 0; f < 64; ++f) {
        float v  = hp[f];
        float vl = v > 0.f ? v : NEG_SLOPE * v;
        s0 += vl * as[f];
        s1 += vl * as[64 + f];
    }
    ss[i] = s0; sd[i] = s1;
}

// scores[e,h] = mean_bv(s_src[bv,src]+s_dst[bv,dst]); per-block per-head max
__global__ void k_edge_scores(const int* __restrict__ ei,
                              const float* __restrict__ ss, const float* __restrict__ sd,
                              float* __restrict__ scores, float* __restrict__ blkmax,
                              int E, int ET, int N) {
    int e = blockIdx.x * 256 + threadIdx.x;
    float loc[4] = {-3.0e38f, -3.0e38f, -3.0e38f, -3.0e38f};
    if (e < ET) {
        int s = (e < E) ? ei[e]     : e - E;       // self-loops appended
        int d = (e < E) ? ei[E + e] : e - E;
        #pragma unroll
        for (int h = 0; h < 4; ++h) {
            float sc = 0.f;
            #pragma unroll
            for (int bv = 0; bv < 4; ++bv)
                sc += ss[(((size_t)bv * N + s) << 2) + h] + sd[(((size_t)bv * N + d) << 2) + h];
            sc *= 0.25f;
            scores[(size_t)e * 4 + h] = sc;
            loc[h] = sc;
        }
    }
    __shared__ float sm[4][256];
    int t = threadIdx.x;
    for (int h = 0; h < 4; ++h) sm[h][t] = loc[h];
    __syncthreads();
    for (int st = 128; st > 0; st >>= 1) {
        if (t < st)
            for (int h = 0; h < 4; ++h) sm[h][t] = fmaxf(sm[h][t], sm[h][t + st]);
        __syncthreads();
    }
    if (t < 4) blkmax[blockIdx.x * 4 + t] = sm[t][0];
}

__global__ void k_reduce_max(const float* __restrict__ bm, int nblk,
                             float* __restrict__ maxs, float* __restrict__ sums) {
    __shared__ float sm[4][256];
    int t = threadIdx.x;
    float loc[4] = {-3.0e38f, -3.0e38f, -3.0e38f, -3.0e38f};
    for (int i = t; i < nblk; i += 256)
        for (int h = 0; h < 4; ++h) loc[h] = fmaxf(loc[h], bm[i * 4 + h]);
    for (int h = 0; h < 4; ++h) sm[h][t] = loc[h];
    __syncthreads();
    for (int st = 128; st > 0; st >>= 1) {
        if (t < st)
            for (int h = 0; h < 4; ++h) sm[h][t] = fmaxf(sm[h][t], sm[h][t + st]);
        __syncthreads();
    }
    if (t < 4) { maxs[t] = sm[t][0]; sums[t] = 0.0f; }
}

__global__ void k_exp_sum(const float* __restrict__ scores, const float* __restrict__ maxs,
                          float* __restrict__ expw, float* __restrict__ sums, int ET) {
    int e = blockIdx.x * 256 + threadIdx.x;
    float loc[4] = {0.f, 0.f, 0.f, 0.f};
    if (e < ET) {
        #pragma unroll
        for (int h = 0; h < 4; ++h) {
            float w = expf(scores[(size_t)e * 4 + h] - maxs[h]);
            expw[(size_t)e * 4 + h] = w;
            loc[h] = w;
        }
    }
    __shared__ float sm[4][256];
    int t = threadIdx.x;
    for (int h = 0; h < 4; ++h) sm[h][t] = loc[h];
    __syncthreads();
    for (int st = 128; st > 0; st >>= 1) {
        if (t < st)
            for (int h = 0; h < 4; ++h) sm[h][t] += sm[h][t + st];
        __syncthreads();
    }
    if (t < 4) atomicAdd(&sums[t], sm[t][0]);
}

__global__ void k_inv_sums(float* __restrict__ sums) {
    if (threadIdx.x < 4) sums[threadIdx.x] = 1.0f / sums[threadIdx.x];
}

// agg[bv,dst,d] += h[bv,src,d] * softmax_weight[e, d>>6]
__global__ void k_scatter(const int* __restrict__ ei, const float* __restrict__ expw,
                          const float* __restrict__ invs, const float* __restrict__ Hf,
                          float* __restrict__ agg, int E, int ET, int N, long total) {
    long i = (long)blockIdx.x * blockDim.x + threadIdx.x;
    if (i >= total) return;
    int d = (int)(i & 255);
    int e = (int)(i >> 8);
    int s  = (e < E) ? ei[e]     : e - E;
    int dd = (e < E) ? ei[E + e] : e - E;
    int h = d >> 6;
    float w = expw[(size_t)e * 4 + h] * invs[h];
    #pragma unroll
    for (int bv = 0; bv < 4; ++bv)
        atomicAdd(&agg[(((size_t)bv * N + dd) << 8) + d],
                  Hf[(((size_t)bv * N + s) << 8) + d] * w);
}

// ---------------- tiny 4-token MHA over views ----------------
// QKV rows are (n*4+v), cols [0..255]=q [256..511]=k [512..767]=v
__global__ void k_attention(const float* __restrict__ QKV, float* __restrict__ O, int N) {
    int tid = blockIdx.x * blockDim.x + threadIdx.x;
    if (tid >= N * 16) return;
    int vq = tid & 3, h = (tid >> 2) & 3, n = tid >> 4;
    const float* base = QKV + (size_t)n * 4 * 768;
    const float* q = base + vq * 768 + h * 64;
    float lg[4];
    #pragma unroll
    for (int vk = 0; vk < 4; ++vk) {
        const float* kp = base + vk * 768 + 256 + h * 64;
        float dt = 0.f;
        #pragma unroll 8
        for (int f = 0; f < 64; ++f) dt += q[f] * kp[f];
        lg[vk] = dt * 0.125f;                      // 1/sqrt(64)
    }
    float m = fmaxf(fmaxf(lg[0], lg[1]), fmaxf(lg[2], lg[3]));
    float a[4], s = 0.f;
    #pragma unroll
    for (int vk = 0; vk < 4; ++vk) { a[vk] = expf(lg[vk] - m); s += a[vk]; }
    float inv = 1.0f / s;
    #pragma unroll
    for (int vk = 0; vk < 4; ++vk) a[vk] *= inv;
    float* o = O + ((size_t)(n * 4 + vq) << 8) + h * 64;
    #pragma unroll 4
    for (int f = 0; f < 64; ++f) {
        float acc = 0.f;
        #pragma unroll
        for (int vk = 0; vk < 4; ++vk)
            acc += a[vk] * base[vk * 768 + 512 + h * 64 + f];
        o[f] = acc;
    }
}

// ---------------- finalize: +out_b +bias, (elu), transpose (n,v)->(v,n) ----
__global__ void k_finalize_l1(const float* __restrict__ mix, const float* __restrict__ outb,
                              const float* __restrict__ bias, bf16* __restrict__ dst,
                              int N, long n) {
    long i = (long)blockIdx.x * blockDim.x + threadIdx.x;
    if (i >= n) return;
    int d = (int)(i & 255);
    int r = (int)(i >> 8);
    int v = r & 3, nd = r >> 2;
    float x = mix[i] + outb[d] + bias[d];
    x = x > 0.f ? x : expf(x) - 1.0f;              // elu between layers
    dst[((size_t)(v * N + nd) << 8) + d] = (bf16)x;   // becomes layer-2 A operand
}

__global__ void k_finalize_l2(const float* __restrict__ mix, const float* __restrict__ outb,
                              const float* __restrict__ bias, float* __restrict__ dst,
                              int N, long n) {
    long i = (long)blockIdx.x * blockDim.x + threadIdx.x;
    if (i >= n) return;
    int d = (int)(i & 255);
    int r = (int)(i >> 8);
    int v = r & 3, nd = r >> 2;
    dst[((size_t)(v * N + nd) << 8) + d] = mix[i] + outb[d] + bias[d];  // [B,V,N,D]
}

// ---------------------------------------------------------------------------

extern "C" void kernel_launch(void* const* d_in, const int* in_sizes, int n_in,
                              void* d_out, int out_size, void* d_ws, size_t ws_size,
                              hipStream_t stream) {
    (void)n_in; (void)out_size; (void)ws_size;
    const int Nn = 10000, Dd = 256, DIN = 64;
    const int E  = in_sizes[1] / 2;                // 128000
    const int ET = E + Nn;                          // + self loops
    const int M  = 4 * Nn;                          // BV * N = 40000 rows

    const float* x  = (const float*)d_in[0];
    const int*   ei = (const int*)d_in[1];
    struct Lw { const float *W, *att, *qw, *qb, *ow, *ob, *bias; int K; };
    Lw L[2] = {
        {(const float*)d_in[2], (const float*)d_in[3], (const float*)d_in[4],
         (const float*)d_in[5], (const float*)d_in[6], (const float*)d_in[7],
         (const float*)d_in[8], DIN},
        {(const float*)d_in[9], (const float*)d_in[10], (const float*)d_in[11],
         (const float*)d_in[12], (const float*)d_in[13], (const float*)d_in[14],
         (const float*)d_in[15], Dd},
    };

    // workspace layout
    char* ws = (char*)d_ws; size_t off = 0;
    auto grab = [&](size_t b) -> char* {
        char* p = ws + off; off = (off + b + 255) & ~(size_t)255; return p;
    };
    float* Hf   = (float*)grab((size_t)M * Dd * 4);       // X@W / out-proj result
    float* AGG  = (float*)grab((size_t)M * Dd * 4);       // edge aggregation
    float* QKV  = (float*)grab((size_t)M * 3 * Dd * 4);   // qkv projections
    float* ATTO = (float*)grab((size_t)M * Dd * 4);       // attention output
    bf16*  BB   = (bf16*)grab((size_t)M * Dd * 2);        // bf16 A staging
    bf16*  WTB  = (bf16*)grab((size_t)3 * Dd * Dd * 2);   // bf16 weight staging
    float* SS   = (float*)grab((size_t)M * 4 * 4);
    float* SD   = (float*)grab((size_t)M * 4 * 4);
    float* SC   = (float*)grab((size_t)ET * 4 * 4);
    float* EW   = (float*)grab((size_t)ET * 4 * 4);
    const int nblkE = (ET + 255) / 256;
    float* BM   = (float*)grab((size_t)nblkE * 4 * 4);
    float* MX   = (float*)grab(64);
    float* SMS  = (float*)grab(64);

    auto cdiv = [](long a, long b) { return (int)((a + b - 1) / b); };
    const int Mtiles = M / 16;                      // 2500
    const int mBlks  = (Mtiles + 7) / 8;            // 8 M-tiles per block

    // X (f32, rows bv*N+n, K=64) -> bf16
    { long c = (long)M * DIN; k_cvt_plain<<<cdiv(c, 256), 256, 0, stream>>>(x, BB, c); }

    for (int l = 0; l < 2; ++l) {
        const int K = L[l].K;
        // h = X @ W   (WMMA, LDS-staged weights)
        { long c = (long)K * Dd;
          k_cvt_wt<<<cdiv(c, 256), 256, 0, stream>>>(L[l].W, WTB, K, Dd, c); }
        k_wmma_gemm<<<mBlks * (Dd / 64), 256, 0, stream>>>(BB, WTB, Hf, nullptr,
                                                           Mtiles, K, Dd);
        // GAT edge softmax (global, per head)
        { long c = (long)M * 4;
          k_att_scores<<<cdiv(c, 256), 256, 0, stream>>>(Hf, L[l].att, SS, SD, c); }
        k_edge_scores<<<nblkE, 256, 0, stream>>>(ei, SS, SD, SC, BM, E, ET, Nn);
        k_reduce_max<<<1, 256, 0, stream>>>(BM, nblkE, MX, SMS);
        k_exp_sum<<<nblkE, 256, 0, stream>>>(SC, MX, EW, SMS, ET);
        k_inv_sums<<<1, 32, 0, stream>>>(SMS);
        // weighted scatter-sum to destination nodes
        { long c = (long)M * Dd; k_zero<<<cdiv(c, 256), 256, 0, stream>>>(AGG, c); }
        { long c = (long)ET * Dd;
          k_scatter<<<cdiv(c, 256), 256, 0, stream>>>(ei, EW, SMS, Hf, AGG, E, ET, Nn, c); }
        // inter-view MHA: qkv proj (WMMA), 4x4 attention, out proj (WMMA)
        { long c = (long)M * Dd; k_cvt_nv<<<cdiv(c, 256), 256, 0, stream>>>(AGG, BB, Nn, c); }
        { long c = (long)3 * Dd * Dd;
          k_cvt_plain<<<cdiv(c, 256), 256, 0, stream>>>(L[l].qw, WTB, c); }
        k_wmma_gemm<<<mBlks * (3 * Dd / 64), 256, 0, stream>>>(BB, WTB, QKV, L[l].qb,
                                                               Mtiles, Dd, 3 * Dd);
        { long c = (long)Nn * 16;
          k_attention<<<cdiv(c, 256), 256, 0, stream>>>(QKV, ATTO, Nn); }
        { long c = (long)M * Dd; k_cvt_plain<<<cdiv(c, 256), 256, 0, stream>>>(ATTO, BB, c); }
        { long c = (long)Dd * Dd;
          k_cvt_plain<<<cdiv(c, 256), 256, 0, stream>>>(L[l].ow, WTB, c); }
        k_wmma_gemm<<<mBlks * (Dd / 64), 256, 0, stream>>>(BB, WTB, Hf, nullptr,
                                                           Mtiles, Dd, Dd);
        // bias add (+elu) and (n,v)->(v,n) transpose
        { long c = (long)M * Dd;
          if (l == 0)
              k_finalize_l1<<<cdiv(c, 256), 256, 0, stream>>>(Hf, L[l].ob, L[l].bias, BB, Nn, c);
          else
              k_finalize_l2<<<cdiv(c, 256), 256, 0, stream>>>(Hf, L[l].ob, L[l].bias,
                                                              (float*)d_out, Nn, c);
        }
    }
}
